// G_MLC_43714177138705
// MI455X (gfx1250) — compile-verified, matrix operand order checked by reading, not run on hardware
//
#include <hip/hip_runtime.h>

// ---------------------------------------------------------------------------
// CDNA5 (gfx1250): all matrix products via v_wmma_f32_16x16x32_bf16 (wave32,
// fp32 accumulate). Heavy GEMMs: 128x128 block tile (8 waves x 8 accs,
// 16 WMMA / wave / 64-K stage); skinny/tailed GEMMs: 64x64 tile.
// Staging: global_load_b128 -> v_cvt_pk_bf16_f32 -> ds_store_b128/b32;
// fragments return as ds_load_b128.
// ---------------------------------------------------------------------------

typedef __bf16 bf16_t;
typedef __attribute__((ext_vector_type(2)))  __bf16 bf16x2;
typedef __attribute__((ext_vector_type(16))) __bf16 bf16x16;
typedef __attribute__((ext_vector_type(8)))  float  floatx8;

#define RB   256      // B
#define RS   64       // S
#define RR   256      // R
#define RD   256      // D
#define RH   4        // H
#define RDH  64       // DH
#define RC   10       // C
#define RK   6        // K

__device__ __forceinline__ floatx8 wmma_bf16(bf16x16 a, bf16x16 b, floatx8 c) {
  return __builtin_amdgcn_wmma_f32_16x16x32_bf16(
      false, a, false, b, (short)0, c, false, false);
}

// 16x32 bf16 fragment from LDS (row-major, leading dim ld). ISA 7.12.2 layout.
__device__ __forceinline__ bf16x16 load_frag(const bf16_t* p, int ld) {
  int lane = threadIdx.x & 31;
  const bf16_t* row = p + (lane & 15) * ld + ((lane >> 4) << 3);
  bf16x16 f;
#pragma unroll
  for (int r = 0; r < 8; ++r) {
    int k = (r < 4) ? (2 * r) : (16 + 2 * (r - 4));
    f[2 * r]     = row[k];
    f[2 * r + 1] = row[k + 1];
  }
  return f;
}

// C/D 16x16 f32 layout: VGPR r -> row r + 8*(lane/16), col = lane%16.
__device__ __forceinline__ void store_acc_lds(float* p, int ld, floatx8 acc, float scale) {
  int lane = threadIdx.x & 31;
  int n  = lane & 15;
  int mb = (lane >> 4) << 3;
#pragma unroll
  for (int r = 0; r < 8; ++r) p[(mb + r) * ld + n] = acc[r] * scale;
}

// Two f32 -> packed bf16x2 in one VGPR (v_cvt_pk_bf16_f32 when available).
__device__ __forceinline__ unsigned pack_bf16(float a, float b) {
#if __has_builtin(__builtin_amdgcn_cvt_pk_bf16_f32)
  bf16x2 h = __builtin_amdgcn_cvt_pk_bf16_f32(a, b);
  union { bf16x2 v; unsigned u; } p;
  p.v = h;
  return p.u;
#else
  union { bf16_t h[2]; unsigned u; } p;
  p.h[0] = (bf16_t)a;
  p.h[1] = (bf16_t)b;
  return p.u;
#endif
}

// 8 floats -> 8 bf16, one 128-bit LDS store (dst must be 16B aligned).
__device__ __forceinline__ void cvt8_store(bf16_t* dst, float4 v0, float4 v1, float s) {
  union { unsigned u[4]; uint4 q; } p;
  p.u[0] = pack_bf16(v0.x * s, v0.y * s);
  p.u[1] = pack_bf16(v0.z * s, v0.w * s);
  p.u[2] = pack_bf16(v1.x * s, v1.y * s);
  p.u[3] = pack_bf16(v1.z * s, v1.w * s);
  *(uint4*)dst = p.q;
}

// ---------------------------------------------------------------------------
// Heavy GEMM: 128x128 block tile, K-step 64. REQUIRES M%128==0, N%128==0,
// K%64==0 (true for all uses: K/V proj, out proj, masked hW1).
// ---------------------------------------------------------------------------
template <bool MASKED>
__global__ __launch_bounds__(256) void gemm128_wmma_kernel(
    const float* __restrict__ A, const float* __restrict__ B,
    const float* __restrict__ bias, const int* __restrict__ rowmask,
    float* __restrict__ C, int Kd, int lda, int ldb, int ldc, int relu)
{
  __shared__ bf16_t sA[128][64];    // 16 KB
  __shared__ bf16_t sBT[128][64];   // 16 KB, transposed: sBT[n][k]

  int tid = threadIdx.x, wave = tid >> 5, lane = tid & 31;
  int m0 = blockIdx.y * 128, n0 = blockIdx.x * 128;
  int wm = (wave >> 2) << 6;        // 0 / 64
  int wn = (wave & 3) << 5;         // 0,32,64,96

  floatx8 acc[4][2];
#pragma unroll
  for (int i = 0; i < 4; ++i)
#pragma unroll
    for (int j = 0; j < 2; ++j)
      acc[i][j] = (floatx8){0.f,0.f,0.f,0.f,0.f,0.f,0.f,0.f};

  for (int k0 = 0; k0 < Kd; k0 += 64) {
    // A tile: 128x64, 8-float chunks -> one ds_store_b128 each
#pragma unroll
    for (int it = 0; it < 4; ++it) {
      int i  = tid + it * 256;
      int r  = i >> 3;
      int c0 = (i & 7) << 3;
      const float* src = A + (size_t)(m0 + r) * lda + k0 + c0;
      float4 v0 = *(const float4*)src;
      float4 v1 = *(const float4*)(src + 4);
      float s = MASKED ? (float)rowmask[(m0 + r) & (RR - 1)] : 1.f;
      cvt8_store(&sA[r][c0], v0, v1, s);
    }
    // B tile transposed: two K rows per chunk -> packed b32 stores
#pragma unroll
    for (int it = 0; it < 2; ++it) {
      int i  = tid + it * 256;
      int kp = i >> 4;                 // 0..31 : k pair
      int nq = (i & 15) << 3;          // 0..120
      const float* s0 = B + (size_t)(k0 + 2 * kp) * ldb + n0 + nq;
      const float* s1 = s0 + ldb;
      float4 a0 = *(const float4*)s0;
      float4 a1 = *(const float4*)(s0 + 4);
      float4 b0 = *(const float4*)s1;
      float4 b1 = *(const float4*)(s1 + 4);
      float va[8] = {a0.x,a0.y,a0.z,a0.w,a1.x,a1.y,a1.z,a1.w};
      float vb[8] = {b0.x,b0.y,b0.z,b0.w,b1.x,b1.y,b1.z,b1.w};
#pragma unroll
      for (int j = 0; j < 8; ++j)
        *(unsigned*)&sBT[nq + j][2 * kp] = pack_bf16(va[j], vb[j]);
    }
    __syncthreads();
#pragma unroll
    for (int kt = 0; kt < 2; ++kt) {
      bf16x16 b0 = load_frag(&sBT[wn +  0][kt * 32], 64);
      bf16x16 b1 = load_frag(&sBT[wn + 16][kt * 32], 64);
#pragma unroll
      for (int mt = 0; mt < 4; ++mt) {
        bf16x16 a = load_frag(&sA[wm + mt * 16][kt * 32], 64);
        acc[mt][0] = wmma_bf16(a, b0, acc[mt][0]);
        acc[mt][1] = wmma_bf16(a, b1, acc[mt][1]);
      }
    }
    __syncthreads();
  }

  int nA = lane & 15, mb = (lane >> 4) << 3;
#pragma unroll
  for (int mt = 0; mt < 4; ++mt) {
#pragma unroll
    for (int t = 0; t < 2; ++t) {
      int gn = n0 + wn + t * 16 + nA;
      float bv = bias ? bias[gn] : 0.f;
#pragma unroll
      for (int r = 0; r < 8; ++r) {
        size_t m = (size_t)(m0 + wm + mt * 16 + mb + r);
        float v = acc[mt][t][r] + bv;
        if (relu) v = fmaxf(v, 0.f);
        C[m * ldc + gn] = v;
      }
    }
  }
}

// ---------------------------------------------------------------------------
// Small GEMM: 64x64 tile, K-step 64, K-tail path (K=2000), optional +=C.
// REQUIRES M%64==0, N%64==0. Used for: rule (x2, acc), Q, hW2 (N=64).
// ---------------------------------------------------------------------------
__global__ __launch_bounds__(256) void gemm64_wmma_kernel(
    const float* __restrict__ A, const float* __restrict__ B,
    const float* __restrict__ bias, float* __restrict__ C,
    int Kd, int lda, int ldb, int ldc, int accumulate)
{
  __shared__ bf16_t sA[64][64];
  __shared__ bf16_t sBT[64][64];

  int tid = threadIdx.x, wave = tid >> 5, lane = tid & 31;
  int m0 = blockIdx.y * 64, n0 = blockIdx.x * 64;
  int wm = wave >> 1;              // 0..3 : 16-row slab
  int wn = wave & 1;               // 0..1 : 32-col slab

  floatx8 acc0 = {0.f,0.f,0.f,0.f,0.f,0.f,0.f,0.f};
  floatx8 acc1 = {0.f,0.f,0.f,0.f,0.f,0.f,0.f,0.f};

  auto stage = [&](int k0, bool tail) {
    // A tile: 64x64, 8-float chunks
#pragma unroll
    for (int it = 0; it < 2; ++it) {
      int i  = tid + it * 256;
      int r  = i >> 3;
      int c0 = (i & 7) << 3;
      const float* src = A + (size_t)(m0 + r) * lda + k0 + c0;
      bf16_t* dst = &sA[r][c0];
      if (!tail) {
        cvt8_store(dst, *(const float4*)src, *(const float4*)(src + 4), 1.f);
      } else {
#pragma unroll
        for (int j = 0; j < 8; ++j) {
          float v = (k0 + c0 + j < Kd) ? src[j] : 0.f;
          dst[j] = (bf16_t)v;
        }
      }
    }
    // B tile transposed: two K rows per chunk -> packed b32 stores
    {
      int kp = tid >> 3;               // 0..31 : k pair
      int nq = (tid & 7) << 3;         // 0..56
      int gk0 = k0 + 2 * kp;
      const float* s0 = B + (size_t)gk0 * ldb + n0 + nq;
      const float* s1 = s0 + ldb;
      float va[8], vb[8];
      if (!tail) {
        float4 a0 = *(const float4*)s0;
        float4 a1 = *(const float4*)(s0 + 4);
        float4 b0 = *(const float4*)s1;
        float4 b1 = *(const float4*)(s1 + 4);
        va[0]=a0.x; va[1]=a0.y; va[2]=a0.z; va[3]=a0.w;
        va[4]=a1.x; va[5]=a1.y; va[6]=a1.z; va[7]=a1.w;
        vb[0]=b0.x; vb[1]=b0.y; vb[2]=b0.z; vb[3]=b0.w;
        vb[4]=b1.x; vb[5]=b1.y; vb[6]=b1.z; vb[7]=b1.w;
      } else {
#pragma unroll
        for (int j = 0; j < 8; ++j) {
          va[j] = (gk0     < Kd) ? s0[j] : 0.f;
          vb[j] = (gk0 + 1 < Kd) ? s1[j] : 0.f;
        }
      }
#pragma unroll
      for (int j = 0; j < 8; ++j)
        *(unsigned*)&sBT[nq + j][2 * kp] = pack_bf16(va[j], vb[j]);
    }
  };

  auto compute = [&]() {
    bf16x16 a0  = load_frag(&sA[wm * 16][0],  64);
    bf16x16 a1  = load_frag(&sA[wm * 16][32], 64);
    bf16x16 b00 = load_frag(&sBT[wn * 32][0],       64);
    bf16x16 b01 = load_frag(&sBT[wn * 32][32],      64);
    bf16x16 b10 = load_frag(&sBT[wn * 32 + 16][0],  64);
    bf16x16 b11 = load_frag(&sBT[wn * 32 + 16][32], 64);
    acc0 = wmma_bf16(a0, b00, acc0);
    acc0 = wmma_bf16(a1, b01, acc0);
    acc1 = wmma_bf16(a0, b10, acc1);
    acc1 = wmma_bf16(a1, b11, acc1);
  };

  int k0 = 0;
  for (; k0 + 64 <= Kd; k0 += 64) {
    stage(k0, false);
    __syncthreads();
    compute();
    __syncthreads();
  }
  if (k0 < Kd) {                   // K tail (2000 = 31*64 + 16)
    stage(k0, true);
    __syncthreads();
    compute();
    __syncthreads();
  }

  int nA = lane & 15, mb = (lane >> 4) << 3;
  int gm0 = m0 + wm * 16 + mb;
#pragma unroll
  for (int t = 0; t < 2; ++t) {
    floatx8 acc = (t == 0) ? acc0 : acc1;
    int gn = n0 + wn * 32 + t * 16 + nA;
    float bv = bias ? bias[gn] : 0.f;
#pragma unroll
    for (int r = 0; r < 8; ++r) {
      size_t m = (size_t)(gm0 + r);
      float v = acc[r] + bv;
      if (accumulate) v += C[m * ldc + gn];
      C[m * ldc + gn] = v;
    }
  }
}

// ---------------------------------------------------------------------------
// Fused attention: per (b,h) and 64-row tile of R:
//   att = softmax(Q Kx^T / sqrt(DH)); emb[.,h] = att @ Vx   (all WMMA)
// ---------------------------------------------------------------------------
__global__ __launch_bounds__(256) void attn_kernel(
    const float* __restrict__ Q,    // [R, D]   (batch-invariant)
    const float* __restrict__ Kx,   // [B, S, D]
    const float* __restrict__ Vx,   // [B, S, D]
    float* __restrict__ emb)        // [B, R, D]
{
  int bh = blockIdx.y;
  int b = bh >> 2;                  // H = 4
  int h = bh & 3;
  int r0 = blockIdx.x * 64;
  int tid = threadIdx.x, wave = tid >> 5, lane = tid & 31;

  __shared__ bf16_t sQ[64][64];
  __shared__ bf16_t sK[64][64];     // Kx rows == B^T layout [s][dh] for Q K^T
  __shared__ float  sAtt[64][64];
  __shared__ bf16_t sAttB[64][64];
  __shared__ bf16_t sVT[64][64];    // [dh][s] == B^T layout for att @ V

  // sQ / sK: contiguous 8-element chunks -> ds_store_b128
#pragma unroll
  for (int it = 0; it < 2; ++it) {
    int i  = tid + it * 256;
    int r  = i >> 3;
    int c0 = (i & 7) << 3;
    const float* qs = &Q [(r0 + r)     * RD + h * RDH + c0];
    const float* ks = &Kx[(b * RS + r) * RD + h * RDH + c0];
    cvt8_store(&sQ[r][c0], *(const float4*)qs, *(const float4*)(qs + 4), 1.f);
    cvt8_store(&sK[r][c0], *(const float4*)ks, *(const float4*)(ks + 4), 1.f);
  }
  // sVT: two V rows per chunk -> packed b32 transposed stores
#pragma unroll
  for (int it = 0; it < 2; ++it) {
    int i  = tid + it * 256;
    int rp = i >> 4;                  // 0..31 : s pair
    int c0 = (i & 15) << 2;           // 0..60
    const float* v0p = &Vx[(b * RS + 2 * rp)     * RD + h * RDH + c0];
    const float* v1p = &Vx[(b * RS + 2 * rp + 1) * RD + h * RDH + c0];
    float4 v0 = *(const float4*)v0p;
    float4 v1 = *(const float4*)v1p;
    *(unsigned*)&sVT[c0 + 0][2 * rp] = pack_bf16(v0.x, v1.x);
    *(unsigned*)&sVT[c0 + 1][2 * rp] = pack_bf16(v0.y, v1.y);
    *(unsigned*)&sVT[c0 + 2][2 * rp] = pack_bf16(v0.z, v1.z);
    *(unsigned*)&sVT[c0 + 3][2 * rp] = pack_bf16(v0.w, v1.w);
  }
  __syncthreads();

  int wm = wave >> 1;               // 0..3 : 16-row tile
  int wn = wave & 1;                // 0..1 : 32-col slab
  {
    floatx8 a0 = {0.f,0.f,0.f,0.f,0.f,0.f,0.f,0.f};
    floatx8 a1 = {0.f,0.f,0.f,0.f,0.f,0.f,0.f,0.f};
#pragma unroll
    for (int kt = 0; kt < 2; ++kt) {
      bf16x16 af = load_frag(&sQ[wm * 16][kt * 32], 64);
      bf16x16 b0 = load_frag(&sK[(wn * 2 + 0) * 16][kt * 32], 64);
      bf16x16 b1 = load_frag(&sK[(wn * 2 + 1) * 16][kt * 32], 64);
      a0 = wmma_bf16(af, b0, a0);
      a1 = wmma_bf16(af, b1, a1);
    }
    store_acc_lds(&sAtt[wm * 16][(wn * 2 + 0) * 16], 64, a0, 0.125f); // 1/sqrt(64)
    store_acc_lds(&sAtt[wm * 16][(wn * 2 + 1) * 16], 64, a1, 0.125f);
  }
  __syncthreads();

  if (tid < 64) {                   // row softmax over S=64
    float mx = -1e30f;
    for (int j = 0; j < 64; ++j) mx = fmaxf(mx, sAtt[tid][j]);
    float s = 0.f;
    for (int j = 0; j < 64; ++j) {
      float e = __expf(sAtt[tid][j] - mx);
      sAtt[tid][j] = e;
      s += e;
    }
    float inv = 1.f / s;
    for (int j = 0; j < 64; j += 2)
      *(unsigned*)&sAttB[tid][j] =
          pack_bf16(sAtt[tid][j] * inv, sAtt[tid][j + 1] * inv);
  }
  __syncthreads();

  {
    floatx8 a0 = {0.f,0.f,0.f,0.f,0.f,0.f,0.f,0.f};
    floatx8 a1 = {0.f,0.f,0.f,0.f,0.f,0.f,0.f,0.f};
#pragma unroll
    for (int kt = 0; kt < 2; ++kt) {
      bf16x16 af = load_frag(&sAttB[wm * 16][kt * 32], 64);
      bf16x16 b0 = load_frag(&sVT[(wn * 2 + 0) * 16][kt * 32], 64);
      bf16x16 b1 = load_frag(&sVT[(wn * 2 + 1) * 16][kt * 32], 64);
      a0 = wmma_bf16(af, b0, a0);
      a1 = wmma_bf16(af, b1, a1);
    }
    int nA = lane & 15, mb = (lane >> 4) << 3;
#pragma unroll
    for (int t = 0; t < 2; ++t) {
      floatx8 acc = (t == 0) ? a0 : a1;
      int dh = (wn * 2 + t) * 16 + nA;
#pragma unroll
      for (int r = 0; r < 8; ++r) {
        int rr = r0 + wm * 16 + mb + r;
        emb[(b * RR + rr) * RD + h * RDH + dh] = acc[r];
      }
    }
  }
}

// ---------------------------------------------------------------------------
// Edge terms: e_src[i] = hW[i,:]·a_s ; e_dst[i] = hW[i,:]·a_d   (float4)
// ---------------------------------------------------------------------------
template <int F>
__global__ __launch_bounds__(256) void edge_kernel(
    const float* __restrict__ hW, const float* __restrict__ a_s,
    const float* __restrict__ a_d, float* __restrict__ e_src,
    float* __restrict__ e_dst)
{
  int idx = blockIdx.x * 256 + threadIdx.x;
  const float* row = hW + (size_t)idx * F;
  float ss = 0.f, sd = 0.f;
#pragma unroll 8
  for (int f = 0; f < F; f += 4) {
    float4 v  = *(const float4*)(row + f);
    float4 as = *(const float4*)(a_s + f);
    float4 ad = *(const float4*)(a_d + f);
    ss += v.x * as.x + v.y * as.y + v.z * as.z + v.w * as.w;
    sd += v.x * ad.x + v.y * ad.y + v.z * ad.z + v.w * ad.w;
  }
  e_src[idx] = ss;
  e_dst[idx] = sd;
}

// ---------------------------------------------------------------------------
// Fused GAT aggregation: per (b, 32-row tile of dst rules):
//   alpha = softmax_j( leakyrelu(e_dst[i]+e_src[j],0.2) + (adj? 0 : -1e9) )
//   out   = alpha @ hW[b] + bias (opt ReLU)           -- alpha@hW via WMMA
// ---------------------------------------------------------------------------
template <int F>
__global__ __launch_bounds__(256) void gat_agg_kernel(
    const float* __restrict__ hW,         // [B*R, F]
    const float* __restrict__ e_dst,      // [B*R]
    const float* __restrict__ e_src,      // [B*R]
    const unsigned char* __restrict__ adj,// [R,R] bool
    const float* __restrict__ bias,       // [F]
    float* __restrict__ outH,             // [B*R, F]
    int relu)
{
  int b  = blockIdx.y;
  int i0 = blockIdx.x * 32;
  int tid = threadIdx.x, wave = tid >> 5, lane = tid & 31;

  __shared__ float  s_alpha[32][256];     // 32 KB
  __shared__ bf16_t s_alpha_bf[32][256];  // 16 KB
  __shared__ bf16_t s_hwT[F][32];         // B^T tile [f][k]
  __shared__ float  s_esrc[256];
  __shared__ float  s_edst[32];
  __shared__ float  s_red[32][8];

  s_esrc[tid] = e_src[b * RR + tid];
  if (tid < 32) s_edst[tid] = e_dst[b * RR + i0 + tid];
  __syncthreads();

  int row = tid >> 3;           // 0..31 dst row in tile
  int sub = tid & 7;            // 8 column chunks of 32
  int ig  = i0 + row;
  float ed = s_edst[row];
  float lmax = -1e30f;
#pragma unroll 4
  for (int jj = 0; jj < 32; ++jj) {
    int j = sub * 32 + jj;
    float v = ed + s_esrc[j];
    v = (v > 0.f) ? v : 0.2f * v;               // leaky_relu(.,0.2)
    v += adj[ig * RR + j] ? 0.f : -1e9f;        // adj bias
    s_alpha[row][j] = v;
    lmax = fmaxf(lmax, v);
  }
  s_red[row][sub] = lmax;
  __syncthreads();
  float rmax = s_red[row][0];
#pragma unroll
  for (int t = 1; t < 8; ++t) rmax = fmaxf(rmax, s_red[row][t]);
  float lsum = 0.f;
#pragma unroll 4
  for (int jj = 0; jj < 32; ++jj) {
    int j = sub * 32 + jj;
    float e = __expf(s_alpha[row][j] - rmax);
    s_alpha[row][j] = e;
    lsum += e;
  }
  __syncthreads();              // all rmax reads done before s_red reuse
  s_red[row][sub] = lsum;
  __syncthreads();
  float rsum = 0.f;
#pragma unroll
  for (int t = 0; t < 8; ++t) rsum += s_red[row][t];
  float inv = 1.f / rsum;
#pragma unroll 4
  for (int jj = 0; jj < 32; jj += 2) {
    int j = sub * 32 + jj;
    *(unsigned*)&s_alpha_bf[row][j] =
        pack_bf16(s_alpha[row][j] * inv, s_alpha[row][j + 1] * inv);
  }
  __syncthreads();

  // out[32,F] = alpha[32,256] @ hW[b][256,F]   (K loop, WMMA)
  constexpr int QPR = F / 4;    // float4 quads per hW row
  int wm = wave & 1;            // 16-row tile (2 tiles cover 32 rows)
  int wq = wave >> 1;           // 0..3
  constexpr int NT = F >> 4;    // 8 (F=128) or 4 (F=64)
  int nt0 = wq, nt1 = wq + 4;
  floatx8 acc0 = {0.f,0.f,0.f,0.f,0.f,0.f,0.f,0.f};
  floatx8 acc1 = {0.f,0.f,0.f,0.f,0.f,0.f,0.f,0.f};

  for (int kt = 0; kt < 8; ++kt) {
    // stage hW^T: two K rows per chunk -> packed b32 transposed stores
#pragma unroll
    for (int it = 0; it < (16 * QPR) / 256; ++it) {
      int i  = tid + it * 256;
      int kp = i / QPR;               // 0..15 : k pair
      int f0 = (i % QPR) * 4;
      const float* r0p = &hW[(size_t)(b * RR + kt * 32 + 2 * kp)     * F + f0];
      const float* r1p = &hW[(size_t)(b * RR + kt * 32 + 2 * kp + 1) * F + f0];
      float4 v0 = *(const float4*)r0p;
      float4 v1 = *(const float4*)r1p;
      *(unsigned*)&s_hwT[f0 + 0][2 * kp] = pack_bf16(v0.x, v1.x);
      *(unsigned*)&s_hwT[f0 + 1][2 * kp] = pack_bf16(v0.y, v1.y);
      *(unsigned*)&s_hwT[f0 + 2][2 * kp] = pack_bf16(v0.z, v1.z);
      *(unsigned*)&s_hwT[f0 + 3][2 * kp] = pack_bf16(v0.w, v1.w);
    }
    __syncthreads();
    bf16x16 af = load_frag(&s_alpha_bf[wm * 16][kt * 32], 256);
    {
      bf16x16 bfr = load_frag(&s_hwT[nt0 * 16][0], 32);
      acc0 = wmma_bf16(af, bfr, acc0);
    }
    if (nt1 < NT) {
      bf16x16 bfr = load_frag(&s_hwT[nt1 * 16][0], 32);
      acc1 = wmma_bf16(af, bfr, acc1);
    }
    __syncthreads();
  }

  int nA = lane & 15, mb = (lane >> 4) << 3;
#pragma unroll
  for (int t = 0; t < 2; ++t) {
    int nt = (t == 0) ? nt0 : nt1;
    if (nt >= NT) continue;
    floatx8 acc = (t == 0) ? acc0 : acc1;
    int gn = nt * 16 + nA;
    float bv = bias[gn];
#pragma unroll
    for (int r = 0; r < 8; ++r) {
      int gi = b * RR + i0 + wm * 16 + mb + r;
      float v = acc[r] + bv;
      if (relu) v = fmaxf(v, 0.f);
      outH[(size_t)gi * F + gn] = v;
    }
  }
}

// ---------------------------------------------------------------------------
// Class head: s[b] = sum_r h2[b,r,:]; logits = s@Wl + R*bl; log_softmax.
// ---------------------------------------------------------------------------
__global__ __launch_bounds__(256) void head_kernel(
    const float* __restrict__ h2,   // [B*R, 64]
    const float* __restrict__ Wl,   // [64, K]
    const float* __restrict__ bl,   // [K]
    float* __restrict__ out, int c) // out[(c*B+b)*K + k]
{
  int b = blockIdx.x, tid = threadIdx.x;
  __shared__ float spart[4][64];
  __shared__ float sacc[64];
  __shared__ float slog[RK];
  int f = tid & 63, chunk = tid >> 6;
  float s = 0.f;
  for (int r = chunk; r < RR; r += 4) s += h2[(b * RR + r) * 64 + f];
  spart[chunk][f] = s;
  __syncthreads();
  if (tid < 64) sacc[tid] = spart[0][tid] + spart[1][tid] + spart[2][tid] + spart[3][tid];
  __syncthreads();
  if (tid < RK) {
    float v = (float)RR * bl[tid];
    for (int ff = 0; ff < 64; ++ff) v += sacc[ff] * Wl[ff * RK + tid];
    slog[tid] = v;
  }
  __syncthreads();
  if (tid == 0) {
    float mx = -1e30f;
    for (int k = 0; k < RK; ++k) mx = fmaxf(mx, slog[k]);
    float se = 0.f;
    for (int k = 0; k < RK; ++k) se += __expf(slog[k] - mx);
    float lse = mx + logf(se);
    for (int k = 0; k < RK; ++k) out[(c * RB + b) * RK + k] = slog[k] - lse;
  }
}

// ---------------------------------------------------------------------------
// Host orchestration
// ---------------------------------------------------------------------------
extern "C" void kernel_launch(void* const* d_in, const int* in_sizes, int n_in,
                              void* d_out, int out_size, void* d_ws, size_t ws_size,
                              hipStream_t stream) {
  (void)in_sizes; (void)n_in; (void)out_size; (void)ws_size;
  const float* vis  = (const float*)d_in[0];
  const float* bas  = (const float*)d_in[1];
  const float* cru  = (const float*)d_in[2];
  const float* Wtb = (const float*)d_in[3];  const float* btb = (const float*)d_in[4];
  const float* Wtk = (const float*)d_in[5];  const float* btk = (const float*)d_in[6];
  const float* Wq  = (const float*)d_in[7];  const float* bq  = (const float*)d_in[8];
  const float* Wk  = (const float*)d_in[9];  const float* bk  = (const float*)d_in[10];
  const float* Wv  = (const float*)d_in[11]; const float* bv  = (const float*)d_in[12];
  const float* Wo  = (const float*)d_in[13]; const float* bo  = (const float*)d_in[14];
  const float* W1  = (const float*)d_in[15]; const float* a1s = (const float*)d_in[16];
  const float* a1d = (const float*)d_in[17]; const float* b1  = (const float*)d_in[18];
  const float* W2  = (const float*)d_in[19]; const float* a2s = (const float*)d_in[20];
  const float* a2d = (const float*)d_in[21]; const float* b2  = (const float*)d_in[22];
  const float* Wl  = (const float*)d_in[23]; const float* bl  = (const float*)d_in[24];
  const unsigned char* adj = (const unsigned char*)d_in[25];
  const int* mask = (const int*)d_in[26];
  float* out = (float*)d_out;

  // workspace layout (floats); big buffers aliased after the attention stage
  float* ws   = (float*)d_ws;
  float* Kb   = ws;                       // [B*S, D]   4,194,304
  float* Vb   = Kb + 4194304;             // [B*S, D]   4,194,304
  float* embA = Vb + 4194304;             // [B*R, D]  16,777,216
  float* emb2 = embA + 16777216;          // [B*R, D]  16,777,216
  float* rule = emb2 + 16777216;          // [R, D]        65,536
  float* Qb   = rule + 65536;             // [R, D]        65,536
  float* eS   = Qb + 65536;               // [B*R]         65,536
  float* eD   = eS + 65536;               // [B*R]         65,536
  float* hW1  = embA;                     // alias (free after emb2)
  float* h1   = embA + 8388608;           // alias
  float* hW2  = Kb;                       // alias (free after attention)
  float* h2   = Vb;                       // alias

  dim3 blk(256);
  auto g64  = [](int M, int N) { return dim3((unsigned)(N / 64),  (unsigned)(M / 64)); };
  auto g128 = [](int M, int N) { return dim3((unsigned)(N / 128), (unsigned)(M / 128)); };

  // rule = basic@Wtb + btb + crucial@Wtk + btk   (K=2000, tiny M -> 64-tile)
  gemm64_wmma_kernel<<<g64(RR, RD), blk, 0, stream>>>(bas, Wtb, btb, rule,
      2000, 2000, RD, RD, 0);
  gemm64_wmma_kernel<<<g64(RR, RD), blk, 0, stream>>>(cru, Wtk, btk, rule,
      2000, 2000, RD, RD, 1);
  // Q (batch-invariant)
  gemm64_wmma_kernel<<<g64(RR, RD), blk, 0, stream>>>(rule, Wq, bq, Qb,
      RD, RD, RD, RD, 0);
  // K, V projections (heavy)
  gemm128_wmma_kernel<false><<<g128(RB * RS, RD), blk, 0, stream>>>(vis, Wk, bk,
      nullptr, Kb, RD, RD, RD, RD, 0);
  gemm128_wmma_kernel<false><<<g128(RB * RS, RD), blk, 0, stream>>>(vis, Wv, bv,
      nullptr, Vb, RD, RD, RD, RD, 0);
  // fused attention -> embA [B,R,D]
  attn_kernel<<<dim3(RR / 64, RB * RH), blk, 0, stream>>>(Qb, Kb, Vb, embA);
  // output projection (heavy)
  gemm128_wmma_kernel<false><<<g128(RB * RR, RD), blk, 0, stream>>>(embA, Wo, bo,
      nullptr, emb2, RD, RD, RD, RD, 0);

  for (int c = 0; c < RC; ++c) {
    // GAT layer 1: masked rows (heavy)
    gemm128_wmma_kernel<true><<<g128(RB * RR, 128), blk, 0, stream>>>(emb2,
        W1 + c * RD * 128, nullptr, mask + c * RR, hW1, RD, RD, 128, 128, 0);
    edge_kernel<128><<<RB * RR / 256, blk, 0, stream>>>(hW1, a1s + c * 128,
        a1d + c * 128, eS, eD);
    gat_agg_kernel<128><<<dim3(RR / 32, RB), blk, 0, stream>>>(hW1, eD, eS, adj,
        b1 + c * 128, h1, 1);
    // GAT layer 2 (N=64 -> 64-tile)
    gemm64_wmma_kernel<<<g64(RB * RR, 64), blk, 0, stream>>>(h1,
        W2 + c * 128 * 64, nullptr, hW2, 128, 128, 64, 64, 0);
    edge_kernel<64><<<RB * RR / 256, blk, 0, stream>>>(hW2, a2s + c * 64,
        a2d + c * 64, eS, eD);
    gat_agg_kernel<64><<<dim3(RR / 32, RB), blk, 0, stream>>>(hW2, eD, eS, adj,
        b2 + c * 64, h2, 0);
    // class head + log_softmax
    head_kernel<<<RB, blk, 0, stream>>>(h2, Wl + c * 64 * RK, bl + c * RK, out, c);
  }
}
